// INTLoRALayer_52037823758634
// MI455X (gfx1250) — 8-bit matrix operand order repaired and checked by reading, compiled, not run on hardware
//
#include <hip/hip_runtime.h>

typedef __attribute__((ext_vector_type(8))) int v8i;
typedef __attribute__((ext_vector_type(4))) int v4i;

#define T_DIM   16384
#define IN_DIM  4096
#define OUT_DIM 4096
#define BM 128
#define BN 128
#define BK 64
#define LDSS 80
#define ABYTES (BM * LDSS)
#define LBUF   (2 * ABYTES)
#define KTILES (IN_DIM / BK)

__device__ __forceinline__ void async_copy_b128(unsigned lds_off, unsigned voff,
                                                unsigned long long sbase) {
  asm volatile("global_load_async_to_lds_b128 %0, %1, %2"
               :: "v"(lds_off), "v"(voff), "s"(sbase)
               : "memory");
}
__device__ __forceinline__ void wait_async0() {
  asm volatile("s_wait_asynccnt 0x0" ::: "memory");
}

__global__ __launch_bounds__(256) void pack_i8_kernel(const int* __restrict__ src,
                                                      signed char* __restrict__ dst,
                                                      int n16) {
  int i = blockIdx.x * blockDim.x + threadIdx.x;
  if (i >= n16) return;
  const v4i* s = (const v4i*)src + (size_t)i * 4;
  v4i a = __builtin_nontemporal_load(s + 0);
  v4i b = __builtin_nontemporal_load(s + 1);
  v4i c = __builtin_nontemporal_load(s + 2);
  v4i d = __builtin_nontemporal_load(s + 3);
  v4i p;
  p.x = (a.x & 255) | ((a.y & 255) << 8) | ((a.z & 255) << 16) | (a.w << 24);
  p.y = (b.x & 255) | ((b.y & 255) << 8) | ((b.z & 255) << 16) | (b.w << 24);
  p.z = (c.x & 255) | ((c.y & 255) << 8) | ((c.z & 255) << 16) | (c.w << 24);
  p.w = (d.x & 255) | ((d.y & 255) << 8) | ((d.z & 255) << 16) | (d.w << 24);
  ((v4i*)dst)[i] = p;
}

__global__ __launch_bounds__(256) void gemm_w8a8_kernel(
    const signed char* __restrict__ x8,
    const signed char* __restrict__ w8,
    const int* __restrict__ bias,
    const float* __restrict__ alpha_p,
    const float* __restrict__ beta_p,
    float* __restrict__ out) {
  __shared__ alignas(16) signed char lds[2 * LBUF];

  const int tid   = threadIdx.x;
  const int lane  = tid & 31;
  const int wid   = tid >> 5;
  const int waveM = wid & 3;
  const int waveN = wid >> 2;
  const int half  = lane >> 4;
  const int l16   = lane & 15;

  const int bm0 = blockIdx.y * BM;
  const int bn0 = blockIdx.x * BN;

  const int s0 = tid, s1 = tid + 256;
  const int r0 = s0 >> 2, o0 = (s0 & 3) * 16;
  const int r1 = s1 >> 2, o1 = (s1 & 3) * 16;

  unsigned va0 = (unsigned)(bm0 + r0) * IN_DIM + o0;
  unsigned va1 = (unsigned)(bm0 + r1) * IN_DIM + o1;
  unsigned vb0 = (unsigned)(bn0 + r0) * IN_DIM + o0;
  unsigned vb1 = (unsigned)(bn0 + r1) * IN_DIM + o1;
  const unsigned long long sa = (unsigned long long)(uintptr_t)x8;
  const unsigned long long sb = (unsigned long long)(uintptr_t)w8;

  const unsigned lds0 = (unsigned)(uintptr_t)&lds[0];
  const unsigned la0 = lds0 + r0 * LDSS + o0;
  const unsigned la1 = lds0 + r1 * LDSS + o1;
  const unsigned lb0 = lds0 + ABYTES + r0 * LDSS + o0;
  const unsigned lb1 = lds0 + ABYTES + r1 * LDSS + o1;

  v8i acc[2][4];
#pragma unroll
  for (int i = 0; i < 2; ++i)
#pragma unroll
    for (int j = 0; j < 4; ++j) acc[i][j] = (v8i)0;

  async_copy_b128(la0, va0, sa);
  async_copy_b128(la1, va1, sa);
  async_copy_b128(lb0, vb0, sb);
  async_copy_b128(lb1, vb1, sb);
  wait_async0();
  __syncthreads();

  for (int kb = 0; kb < KTILES; ++kb) {
    const int cur = kb & 1;
    const unsigned cbuf = cur * LBUF;
    const unsigned nbuf = (cur ^ 1) * LBUF;

    if (kb + 1 < KTILES) {
      va0 += BK; va1 += BK; vb0 += BK; vb1 += BK;
      async_copy_b128(la0 + nbuf, va0, sa);
      async_copy_b128(la1 + nbuf, va1, sa);
      async_copy_b128(lb0 + nbuf, vb0, sb);
      async_copy_b128(lb1 + nbuf, vb1, sb);
    }

    const signed char* abase = &lds[cbuf];
    const signed char* bbase = &lds[cbuf + ABYTES];

    v8i afrag[2];
#pragma unroll
    for (int fm = 0; fm < 2; ++fm) {
      const signed char* pa = abase + (waveM * 32 + fm * 16 + l16) * LDSS + half * 8;
      int2 t0 = *(const int2*)(pa + 0);
      int2 t1 = *(const int2*)(pa + 16);
      int2 t2 = *(const int2*)(pa + 32);
      int2 t3 = *(const int2*)(pa + 48);
      afrag[fm][0] = t0.x; afrag[fm][1] = t0.y;
      afrag[fm][2] = t1.x; afrag[fm][3] = t1.y;
      afrag[fm][4] = t2.x; afrag[fm][5] = t2.y;
      afrag[fm][6] = t3.x; afrag[fm][7] = t3.y;
    }

    v8i bfrag[4];
#pragma unroll
    for (int fn = 0; fn < 4; ++fn) {
      const signed char* pb = bbase + (waveN * 64 + fn * 16 + l16) * LDSS + half * 8;
      int2 u0 = *(const int2*)(pb + 0);
      int2 u1 = *(const int2*)(pb + 16);
      int2 u2 = *(const int2*)(pb + 32);
      int2 u3 = *(const int2*)(pb + 48);
      bfrag[fn][0] = u0.x; bfrag[fn][1] = u0.y;
      bfrag[fn][2] = u1.x; bfrag[fn][3] = u1.y;
      bfrag[fn][4] = u2.x; bfrag[fn][5] = u2.y;
      bfrag[fn][6] = u3.x; bfrag[fn][7] = u3.y;
    }

#pragma unroll
    for (int fm = 0; fm < 2; ++fm)
#pragma unroll
      for (int fn = 0; fn < 4; ++fn)
        acc[fm][fn] = __builtin_amdgcn_wmma_i32_16x16x64_iu8(
             true, afrag[fm],  true, bfrag[fn],
            acc[fm][fn],  false,  false);

    if (kb + 1 < KTILES) wait_async0();
    __syncthreads();
  }

  const float alpha = alpha_p[0];
  const float beta  = beta_p[0];
#pragma unroll
  for (int fn = 0; fn < 4; ++fn) {
    const int gcol = bn0 + waveN * 64 + fn * 16 + l16;
    const float bb = beta * (float)bias[gcol];
#pragma unroll
    for (int fm = 0; fm < 2; ++fm) {
      const int grow = bm0 + waveM * 32 + fm * 16 + half * 8;
#pragma unroll
      for (int j = 0; j < 8; ++j) {
        float y = (float)acc[fm][fn][j] * alpha + bb;
        y = rintf(y);
        y = fminf(fmaxf(y, -128.0f), 127.0f);
        __builtin_nontemporal_store(y, &out[(size_t)(grow + j) * OUT_DIM + gcol]);
      }
    }
  }
}

extern "C" void kernel_launch(void* const* d_in, const int* in_sizes, int n_in,
                              void* d_out, int out_size, void* d_ws, size_t ws_size,
                              hipStream_t stream) {
  const int*   x      = (const int*)d_in[0];
  const int*   weight = (const int*)d_in[1];
  const int*   bias   = (const int*)d_in[2];
  const float* alpha  = (const float*)d_in[3];
  const float* beta   = (const float*)d_in[4];
  float* out = (float*)d_out;

  signed char* x8 = (signed char*)d_ws;
  signed char* w8 = x8 + (size_t)T_DIM * IN_DIM;

  const int nx16 = T_DIM * IN_DIM / 16;
  const int nw16 = OUT_DIM * IN_DIM / 16;
  pack_i8_kernel<<<nx16 / 256, 256, 0, stream>>>(x, x8, nx16);
  pack_i8_kernel<<<nw16 / 256, 256, 0, stream>>>(weight, w8, nw16);

  dim3 grid(OUT_DIM / BN, T_DIM / BM);
  gemm_w8a8_kernel<<<grid, 256, 0, stream>>>(x8, w8, bias, alpha, beta, out);
}
